// GraphAttentionLayer_47029891891227
// MI455X (gfx1250) — matrix operand order repaired and checked by reading, not compiled, not run on hardware
//
#include <hip/hip_runtime.h>

typedef __attribute__((ext_vector_type(16))) _Float16 v16h;
typedef __attribute__((ext_vector_type(8)))  _Float16 v8h;
typedef __attribute__((ext_vector_type(8)))  float    v8f;
typedef __attribute__((ext_vector_type(4)))  float    v4f;
typedef __attribute__((ext_vector_type(4)))  int      v4i;
typedef __attribute__((ext_vector_type(8)))  int      v8i;
typedef __attribute__((ext_vector_type(4)))  unsigned int v4u;

#define ALPHA_   0.2f
#define LOG2E_   1.4426950408889634f
#define B_       8
#define N_       2048
#define FIN_     256
#define FOUT_    64
#define NEG_INF_ (-__builtin_inff())

// V tile in LDS: 64 feature-rows x 32 nodes f16 (64B) + 16B TDM pad -> 80B rows
#define VROW_   80
#define VTILE_  (64 * VROW_)    // 5120 B
// adj tile in LDS: 64 node-rows x 32 cols i32 (128B) + 16B TDM pad -> 144B rows
#define AROW_   144
#define ATILE_  (64 * AROW_)    // 9216 B

// ---------------------------------------------------------------------------
// TDM: issue a 2D tensor_load_to_lds.  D# bit layout per CDNA5 ISA ch.7/8:
//   group0: count=1 | lds_addr | global_addr[56:0] | type=2
//   group1: data_size, pad_enable, pad_interval, pad_amount, tensor dims,
//           tile dims, dim0 stride
//   groups 2/3 (+ trailing v8i): zero (2D tile)
// pad_interval code: 3 = every 16 DW (64B), 4 = every 32 DW (128B)
// pad_amount   code: 3 = 4 DW (16B)
// ---------------------------------------------------------------------------
__device__ __forceinline__ void tdm_load_2d(unsigned lds_off, const void* gptr,
                                            unsigned data_size_code,
                                            unsigned tile_d0, unsigned tile_d1,
                                            unsigned long long d0_stride,
                                            unsigned pad_interval,
                                            unsigned pad_amount) {
  const unsigned long long ga = (unsigned long long)gptr;
  v4u g0 = {1u,                                     // count=1, user mode
            lds_off,                                // [63:32] lds addr
            (unsigned)ga,                           // [95:64] ga[31:0]
            (unsigned)((ga >> 32) & 0x01FFFFFFu) | 0x80000000u};  // ga[56:32] | type=2
  v8i g1;
  g1[0] = (int)((data_size_code << 16) | (1u << 20) |
                (pad_interval << 22) | (pad_amount << 25));
  g1[1] = (int)((tile_d0 & 0xFFFFu) << 16);         // tensor_dim0[15:0] (== tile_d0)
  g1[2] = (int)(((tile_d0 >> 16) & 0xFFFFu) | ((tile_d1 & 0xFFFFu) << 16)); // dim0 hi, dim1 lo
  g1[3] = (int)(((tile_d1 >> 16) & 0xFFFFu) | ((tile_d0 & 0xFFFFu) << 16)); // dim1 hi, tile_dim0
  g1[4] = (int)(tile_d1 & 0xFFFFu);                 // tile_dim1, tile_dim2=0
  g1[5] = (int)(unsigned)(d0_stride & 0xFFFFFFFFu); // tensor_dim0_stride[31:0]
  g1[6] = (int)(unsigned)((d0_stride >> 32) & 0xFFFFu);
  g1[7] = 0;
  v4i z4 = {0, 0, 0, 0};
  v8i z8 = {0, 0, 0, 0, 0, 0, 0, 0};
  __builtin_amdgcn_tensor_load_to_lds(g0, g1, z4, z4, z8, 0);
}

// ---------------------------------------------------------------------------
// Kernel 1: WhT[b][o][n] (f16, feature-major) = sum_f h[b][n][f] * W[o][f]
// One 16-row tile per wave via v_wmma_f32_16x16x32_f16.
// ---------------------------------------------------------------------------
__global__ __launch_bounds__(256)
void gat_wh_kernel(const float* __restrict__ h, const float* __restrict__ W,
                   _Float16* __restrict__ WhT) {
  const int lane = threadIdx.x & 31;
  const int wave = threadIdx.x >> 5;
  const int lm   = lane & 15;
  const int hi   = lane >> 4;

  const int  tile    = blockIdx.x * 8 + wave;
  const long rowbase = (long)tile * 16;
  const int  b       = (int)(rowbase >> 11);
  const int  nlocal  = (int)(rowbase & (N_ - 1));

  v8f acc[4];
#pragma unroll
  for (int nt = 0; nt < 4; ++nt) acc[nt] = (v8f){};

  const float* hrow = h + (rowbase + lm) * FIN_;

  for (int kk = 0; kk < FIN_; kk += 32) {
    const float* pa = hrow + kk + hi * 8;
    v4f a0 = *(const v4f*)(pa + 0);
    v4f a1 = *(const v4f*)(pa + 4);
    v4f a2 = *(const v4f*)(pa + 16);
    v4f a3 = *(const v4f*)(pa + 20);
    v16h A;
#pragma unroll
    for (int t = 0; t < 4; ++t) {
      A[t]      = (_Float16)a0[t];
      A[4 + t]  = (_Float16)a1[t];
      A[8 + t]  = (_Float16)a2[t];
      A[12 + t] = (_Float16)a3[t];
    }
#pragma unroll
    for (int nt = 0; nt < 4; ++nt) {
      const float* pw = W + (nt * 16 + lm) * FIN_ + kk + hi * 8;
      v4f w0 = *(const v4f*)(pw + 0);
      v4f w1 = *(const v4f*)(pw + 4);
      v4f w2 = *(const v4f*)(pw + 16);
      v4f w3 = *(const v4f*)(pw + 20);
      v16h Bf;
#pragma unroll
      for (int t = 0; t < 4; ++t) {
        Bf[t]      = (_Float16)w0[t];
        Bf[4 + t]  = (_Float16)w1[t];
        Bf[8 + t]  = (_Float16)w2[t];
        Bf[12 + t] = (_Float16)w3[t];
      }
      acc[nt] = __builtin_amdgcn_wmma_f32_16x16x32_f16(
          false, A, false, Bf, (short)0, acc[nt], false, false);
    }
  }

#pragma unroll
  for (int nt = 0; nt < 4; ++nt) {
    v8h s;
#pragma unroll
    for (int i = 0; i < 8; ++i) s[i] = (_Float16)acc[nt][i];
    const int col = nt * 16 + lm;
    _Float16* dst = WhT + (((long)(b * FOUT_ + col)) << 11) + nlocal + hi * 8;
    *(v8h*)dst = s;
  }
}

// ---------------------------------------------------------------------------
// Kernel 2: e_i/e_j = Wh . a, pre-scaled by log2(e) for exp2-domain softmax
// (valid: softmax is base-invariant, leaky_relu is positively homogeneous)
// ---------------------------------------------------------------------------
__global__ __launch_bounds__(256)
void gat_e_kernel(const _Float16* __restrict__ WhT, const float* __restrict__ a,
                  float* __restrict__ ei, float* __restrict__ ej) {
  const int t = blockIdx.x * 256 + threadIdx.x;
  const int b = t >> 11;
  const int n = t & (N_ - 1);
  const _Float16* base = WhT + (((long)b * FOUT_) << 11) + n;
  float si = 0.f, sj = 0.f;
#pragma unroll 8
  for (int o = 0; o < FOUT_; ++o) {
    float w = (float)base[(long)o << 11];
    si += w * a[o];
    sj += w * a[FOUT_ + o];
  }
  ei[t] = si * LOG2E_;
  ej[t] = sj * LOG2E_;
}

// ---------------------------------------------------------------------------
// Kernel 3: flash-attention style, TDM-fed.  Wave0 streams adj (64x32 i32)
// and V (64x32 f16) tiles into double-buffered LDS via tensor_load_to_lds;
// all 4 waves consume from LDS, do exp2-domain online softmax, and
// accumulate P(16x32 f16) x V(32x64 f16) with v_wmma_f32_16x16x32_f16.
// ---------------------------------------------------------------------------
__global__ __launch_bounds__(128)
void gat_attn_kernel(const int* __restrict__ adj, const _Float16* __restrict__ WhT,
                     const float* __restrict__ ei, const float* __restrict__ ej,
                     float* __restrict__ out) {
  __shared__ __align__(128) unsigned char ldsV[2][VTILE_];
  __shared__ __align__(128) unsigned char ldsA[2][ATILE_];

  const int lane = threadIdx.x & 31;
  const int wave = threadIdx.x >> 5;
  const int lm   = lane & 15;
  const int hi   = lane >> 4;

  const int b       = blockIdx.x >> 5;
  const int rowtile = blockIdx.x & 31;
  const int rowbase = rowtile * 64 + wave * 16;
  const int myrow   = rowbase + lm;

  const float eiv = ei[b * N_ + myrow];
  const float* ejb = ej + b * N_;
  const _Float16* whb = WhT + (((long)b * FOUT_) << 11);
  const int* adjblk = adj + (((long)(b * N_ + rowtile * 64)) << 11);

  float m = NEG_INF_;
  float l = 0.f;
  v8f acc[4];
#pragma unroll
  for (int nt = 0; nt < 4; ++nt) acc[nt] = (v8f){};

  // ---- prologue: wave0 issues TDM loads for tile 0
  if (wave == 0) {
    tdm_load_2d((unsigned)(size_t)&ldsV[0][0], whb, /*2B*/1, 32, 64,
                (unsigned long long)N_, /*64B*/3, /*16B*/3);
    tdm_load_2d((unsigned)(size_t)&ldsA[0][0], adjblk, /*4B*/2, 32, 64,
                (unsigned long long)N_, /*128B*/4, /*16B*/3);
  }

  for (int ct = 0; ct < N_; ct += 32) {
    const int ibuf = (ct >> 5) & 1;

    if (wave == 0) {
      if (ct + 32 < N_) {
        tdm_load_2d((unsigned)(size_t)&ldsV[ibuf ^ 1][0], whb + (ct + 32),
                    1, 32, 64, (unsigned long long)N_, 3, 3);
        tdm_load_2d((unsigned)(size_t)&ldsA[ibuf ^ 1][0], adjblk + (ct + 32),
                    2, 32, 64, (unsigned long long)N_, 4, 3);
        __builtin_amdgcn_s_wait_tensorcnt((short)2);  // current tile done
      } else {
        __builtin_amdgcn_s_wait_tensorcnt((short)0);
      }
    }
    __syncthreads();

    // ---- adjacency from LDS (rows padded to 144B: conflict-free b128 reads)
    const unsigned char* pr = &ldsA[ibuf][(wave * 16 + lm) * AROW_ + hi * 32];
    v4i m0 = *(const v4i*)(pr + 0);
    v4i m1 = *(const v4i*)(pr + 16);
    v4i m2 = *(const v4i*)(pr + 64);
    v4i m3 = *(const v4i*)(pr + 80);

    const int c1 = ct + hi * 8;
    v4f j0 = *(const v4f*)(ejb + c1);
    v4f j1 = *(const v4f*)(ejb + c1 + 4);
    v4f j2 = *(const v4f*)(ejb + c1 + 16);
    v4f j3 = *(const v4f*)(ejb + c1 + 20);

    float ev[16];
    int   ad[16];
#pragma unroll
    for (int t = 0; t < 4; ++t) {
      ad[t] = m0[t];  ad[4 + t] = m1[t];  ad[8 + t] = m2[t];  ad[12 + t] = m3[t];
      ev[t] = j0[t];  ev[4 + t] = j1[t];  ev[8 + t] = j2[t];  ev[12 + t] = j3[t];
    }

    float tmax = NEG_INF_;
#pragma unroll
    for (int t = 0; t < 16; ++t) {
      float e = eiv + ev[t];
      e = (e > 0.f) ? e : (ALPHA_ * e);     // leaky_relu (log2-scaled domain)
      if (ad[t] == 0) e = NEG_INF_;
      ev[t] = e;
      tmax = fmaxf(tmax, e);
    }
    tmax = fmaxf(tmax, __shfl_xor(tmax, 16));

    const float mnew = fmaxf(m, tmax);
    const float corr = (m == NEG_INF_) ? 0.f : exp2f(m - mnew);

    float ps = 0.f;
    v16h P;
#pragma unroll
    for (int t = 0; t < 16; ++t) {
      float p = (ad[t] == 0) ? 0.f : exp2f(ev[t] - mnew);
      ps += p;
      P[t] = (_Float16)p;
    }
    ps += __shfl_xor(ps, 16);
    l = l * corr + ps;

    // ---- rescale accumulator only if some row's max advanced
    if (__any(mnew > m)) {
#pragma unroll
      for (int i = 0; i < 8; ++i) {
        const float cr = __shfl(corr, i + 8 * hi);
#pragma unroll
        for (int nt = 0; nt < 4; ++nt) acc[nt][i] *= cr;
      }
    }
    m = mnew;

    // ---- P x V from LDS V tile (rows padded to 80B: conflict-free reads)
#pragma unroll
    for (int nt = 0; nt < 4; ++nt) {
      const unsigned char* pv = &ldsV[ibuf][(nt * 16 + lm) * VROW_ + hi * 16];
      v8h x0 = *(const v8h*)(pv + 0);
      v8h x1 = *(const v8h*)(pv + 32);
      v16h V;
#pragma unroll
      for (int t = 0; t < 8; ++t) { V[t] = x0[t]; V[8 + t] = x1[t]; }
      acc[nt] = __builtin_amdgcn_wmma_f32_16x16x32_f16(
          false, P, false, V, (short)0, acc[nt], false, false);
    }
    __syncthreads();   // everyone done with buffers before next TDM overwrite
  }

  const float inv = (l > 0.f) ? (1.f / l) : 0.f;   // l==0 -> nan_to_num(0)
#pragma unroll
  for (int i = 0; i < 8; ++i) {
    const float ivr = __shfl(inv, i + 8 * hi);
    const int row = rowbase + i + 8 * hi;
#pragma unroll
    for (int nt = 0; nt < 4; ++nt) {
      out[((long)(b * N_ + row)) * FOUT_ + nt * 16 + lm] = acc[nt][i] * ivr;
    }
  }
}

// ---------------------------------------------------------------------------
extern "C" void kernel_launch(void* const* d_in, const int* in_sizes, int n_in,
                              void* d_out, int out_size, void* d_ws, size_t ws_size,
                              hipStream_t stream) {
  const float* h   = (const float*)d_in[0];   // (8, 2048, 256) f32
  const int*   adj = (const int*)d_in[1];     // (8, 2048, 2048) i32
  const float* W   = (const float*)d_in[2];   // (64, 256) f32
  const float* a   = (const float*)d_in[3];   // (1, 128) f32
  float* out = (float*)d_out;                 // (8, 2048, 64) f32

  char* ws = (char*)d_ws;
  _Float16* WhT = (_Float16*)ws;                                   // 2 MB
  float* ei = (float*)(ws + (size_t)B_ * FOUT_ * N_ * sizeof(_Float16));
  float* ej = ei + B_ * N_;                                        // 64 KB each

  gat_wh_kernel<<<128, 256, 0, stream>>>(h, W, WhT);
  gat_e_kernel<<<(B_ * N_) / 256, 256, 0, stream>>>(WhT, a, ei, ej);
  gat_attn_kernel<<<B_ * 32, 128, 0, stream>>>(adj, WhT, ei, ej, out);
}
